// SelfAttention_50105088475526
// MI455X (gfx1250) — matrix operand order repaired and checked by reading, not compiled, not run on hardware
//
#include <hip/hip_runtime.h>

// ---------------------------------------------------------------------------
// Types for CDNA5 WMMA (wave32): v16bf A/B fragments, v8f f32 accumulator
// ---------------------------------------------------------------------------
typedef __attribute__((ext_vector_type(16))) __bf16 v16bf;
typedef __attribute__((ext_vector_type(8)))  float  v8f;
typedef __attribute__((ext_vector_type(4)))  int    v4i;

union FragAB { uint4 q[2]; v16bf v; };

// gfx1250 async global->LDS copy machinery (probe via __has_builtin; fall back
// to synchronous VGPR-bounce copies if the toolchain lacks the builtins).
#if defined(__gfx1250__) &&                                                    \
    __has_builtin(__builtin_amdgcn_global_load_async_to_lds_b128) &&          \
    __has_builtin(__builtin_amdgcn_s_wait_asynccnt)
#define ASYNC_LDS 1
#else
#define ASYNC_LDS 0
#endif

typedef __attribute__((address_space(1))) v4i* gptr_v4i;
typedef __attribute__((address_space(3))) v4i* lptr_v4i;

static __device__ __forceinline__ void cp16_lds(void* l, const void* g) {
#if ASYNC_LDS
    __builtin_amdgcn_global_load_async_to_lds_b128(
        (gptr_v4i)(g), (lptr_v4i)(l), 0, 0);
#else
    *(uint4*)l = *(const uint4*)g;
#endif
}

static __device__ __forceinline__ unsigned short f2bf(float f) {
    unsigned u = __builtin_bit_cast(unsigned, f);
    unsigned r = u + 0x7FFFu + ((u >> 16) & 1u);   // round-to-nearest-even
    return (unsigned short)(r >> 16);
}

// A fragment (16x32 bf16, M=row K=col) from row-major bf16 matrix, row stride ld.
// Lane L: row = L%16; elems 0..7 -> K = h*8+0..7 ; elems 8..15 -> K = 16+h*8+0..7
static __device__ __forceinline__ v16bf load_a(const unsigned short* base, int ld,
                                               int m0, int k0) {
    int lane = threadIdx.x & 31;
    int row  = m0 + (lane & 15);
    int h    = lane >> 4;
    const uint4* p = (const uint4*)(base + (long long)row * ld + k0 + h * 8);
    FragAB f; f.q[0] = p[0]; f.q[1] = p[2];
    return f.v;
}

// B fragment (32x16 bf16, K x N). Source memory holds element (k, n) at
// base[n*ld + k].  Lane L: col = L%16; elems 0..7 -> K = k0 + h*8 + e ; elems 8..15 -> K = k0 + 16 + h*8 + (e-8)
static __device__ __forceinline__ v16bf load_b(const unsigned short* base, int ld,
                                               int n0, int k0) {
    int lane = threadIdx.x & 31;
    int col  = n0 + (lane & 15);
    int h    = lane >> 4;
    const uint4* p = (const uint4*)(base + (long long)col * ld + k0 + h * 8);
    FragAB f; f.q[0] = p[0]; f.q[1] = p[2];
    return f.v;
}

static __device__ __forceinline__ v8f wmma_bf16(v16bf a, v16bf b, v8f c) {
    return __builtin_amdgcn_wmma_f32_16x16x32_bf16(false, a, false, b,
                                                   (short)0, c, false, false);
}

// ---------------------------------------------------------------------------
// Pack kernels: transpose x to pixel-major bf16; convert weights to bf16
// ---------------------------------------------------------------------------
__global__ void pack_x_kernel(const float* __restrict__ x,
                              unsigned short* __restrict__ xb,
                              int B, int C, int N) {
    long long idx = (long long)blockIdx.x * blockDim.x + threadIdx.x;
    long long total = (long long)B * N * C;
    if (idx >= total) return;
    int c = (int)(idx % C);
    long long bn = idx / C;
    int n = (int)(bn % N);
    int b = (int)(bn / N);
    xb[idx] = f2bf(x[((long long)b * C + c) * N + n]);   // xb[b][n][c]
}

__global__ void pack_w_kernel(const float* __restrict__ Wq,
                              const float* __restrict__ Wk,
                              const float* __restrict__ Wv,
                              const float* __restrict__ Wo,
                              unsigned short* __restrict__ Wcatb,
                              unsigned short* __restrict__ Wob,
                              int KC, int VC, int C) {
    int idx  = blockIdx.x * blockDim.x + threadIdx.x;
    int ncat = 3 * KC * C;
    int nwo  = C * VC;
    if (idx < ncat) {
        int seg = idx / (KC * C);
        int off = idx % (KC * C);
        const float* W = (seg == 0) ? Wq : ((seg == 1) ? Wk : Wv);
        Wcatb[idx] = f2bf(W[off]);                        // [384][C]
    } else if (idx < ncat + nwo) {
        int off = idx - ncat;
        Wob[off] = f2bf(Wo[off]);                         // [C][VC]
    }
}

// ---------------------------------------------------------------------------
// QKV projection: per-wave 16(channel) x 64(pixel) tile.
// ---------------------------------------------------------------------------
__global__ __launch_bounds__(128) void qkv_kernel(
    const unsigned short* __restrict__ xb,    // [B][N][C]
    const unsigned short* __restrict__ Wcatb, // [3*KC][C]
    const float* __restrict__ bq, const float* __restrict__ bk,
    const float* __restrict__ bv,
    unsigned short* __restrict__ Qb,          // [B][N][KC]
    unsigned short* __restrict__ Kb,          // [B][N][KC]
    unsigned short* __restrict__ Vt,          // [B][VC][N]
    int B, int C, int N, int KC, int VC)
{
    const int CH = 3 * KC;
    int wave = (blockIdx.x * blockDim.x + threadIdx.x) >> 5;
    int lane = threadIdx.x & 31;
    int nTilesN = N / 64;
    int nTilesC = CH / 16;
    int nt  = wave % nTilesN;
    int tmp = wave / nTilesN;
    int ct  = tmp % nTilesC;
    int b   = tmp / nTilesC;
    if (b >= B) return;
    int ch0 = ct * 16;
    int n0  = nt * 64;
    int h   = lane >> 4;

    v8f acc[4];
    for (int r = 0; r < 8; ++r) {
        int cc = ch0 + r + 8 * h;
        float bias = (cc < KC) ? bq[cc] : ((cc < 2 * KC) ? bk[cc - KC] : bv[cc - 2 * KC]);
        for (int t = 0; t < 4; ++t) acc[t][r] = bias;
    }

    const unsigned short* xrow = xb + (long long)b * N * C;
    for (int k0 = 0; k0 < C; k0 += 32) {
        v16bf a = load_a(Wcatb, C, ch0, k0);
        for (int t = 0; t < 4; ++t) {
            v16bf bf = load_b(xrow, C, n0 + t * 16, k0);
            acc[t] = wmma_bf16(a, bf, acc[t]);
        }
    }

    for (int t = 0; t < 4; ++t) {
        for (int r = 0; r < 8; ++r) {
            int cc = ch0 + r + 8 * h;
            int n  = n0 + t * 16 + (lane & 15);
            unsigned short val = f2bf(acc[t][r]);
            if (cc < KC)          Qb[((long long)b * N + n) * KC + cc] = val;
            else if (cc < 2 * KC) Kb[((long long)b * N + n) * KC + (cc - KC)] = val;
            else                  Vt[((long long)b * VC + (cc - 2 * KC)) * N + n] = val;
        }
    }
}

// ---------------------------------------------------------------------------
// Flash attention with block-shared, double-buffered LDS K/V tiles.
// Block = 4 waves = 64 query rows of one batch; j-loop over all 4096 keys.
// K/V tiles staged via async global->LDS DMA (ASYNCcnt) when available.
// ---------------------------------------------------------------------------
__global__ __launch_bounds__(128) void attn_kernel(
    const unsigned short* __restrict__ Qb,  // [B][N][KC]
    const unsigned short* __restrict__ Kb,  // [B][N][KC]
    const unsigned short* __restrict__ Vt,  // [B][VC][N]
    unsigned short* __restrict__ Ob,        // [B][N][VC]
    int B, int N, int KC, int VC)
{
    __shared__ unsigned short sK[2][32 * 128];  // [j-local][kc]   8 KB x2
    __shared__ unsigned short sV[2][128 * 32];  // [vc][j-local]   8 KB x2
    __shared__ unsigned short plds[4][16 * 32]; // per-wave P staging

    int tid    = threadIdx.x;
    int wlocal = tid >> 5;
    int lane   = tid & 31;
    int nGrpN  = N / 64;                 // 64 blocks per batch
    int b      = blockIdx.x / nGrpN;
    int itb    = blockIdx.x % nGrpN;
    int i0     = (itb * 4 + wlocal) * 16;
    int h      = lane >> 4;

    const unsigned short* Q = Qb + (long long)b * N * KC;
    const unsigned short* K = Kb + (long long)b * N * KC;
    const unsigned short* V = Vt + (long long)b * VC * N;

    // cooperative stage of one 32-key K tile (contiguous 8KB) + V tile
    auto stage = [&](int buf, int j0) {
        const uint4* gk = (const uint4*)(K + (long long)j0 * KC);
        uint4*       lk = (uint4*)sK[buf];
        uint4*       lv = (uint4*)sV[buf];
#pragma unroll
        for (int i = 0; i < 4; ++i) {
            int idx = tid + 128 * i;                 // 512 x 16B chunks
            cp16_lds(lk + idx, gk + idx);            // K rows contiguous
        }
#pragma unroll
        for (int i = 0; i < 4; ++i) {
            int idx = tid + 128 * i;                 // 512 chunks
            int row = idx >> 2;                      // vc row (0..127)
            int c   = idx & 3;                       // 16B chunk in row
            cp16_lds(lv + idx, (const uint4*)(V + (long long)row * N + j0) + c);
        }
    };

    v16bf qa[4];
#pragma unroll
    for (int kk = 0; kk < 4; ++kk) qa[kk] = load_a(Q, KC, i0, kk * 32);

    v8f acc[8];
#pragma unroll
    for (int t = 0; t < 8; ++t)
        for (int r = 0; r < 8; ++r) acc[t][r] = 0.f;
    float mrow[8], lrow[8];
#pragma unroll
    for (int r = 0; r < 8; ++r) { mrow[r] = -3.0e38f; lrow[r] = 0.f; }

    unsigned short* myP = plds[wlocal];
    const float L2E = 1.44269504088896f;

    stage(0, 0);
    int cur = 0;
    for (int j0 = 0; j0 < N; j0 += 32) {
        int nxt = cur ^ 1;
        bool more = (j0 + 32 < N);
        if (more) stage(nxt, j0 + 32);
        if (more && (lane & 15) == 0)   // warm L2 for the tile after next
            __builtin_prefetch(K + (long long)(j0 + 64) * KC, 0, 2);
#if ASYNC_LDS
        if (more) __builtin_amdgcn_s_wait_asynccnt(8);   // cur tile complete
        else      __builtin_amdgcn_s_wait_asynccnt(0);
#endif
        __syncthreads();                 // cur tile visible to all waves

        const unsigned short* kTile = sK[cur];
        const unsigned short* vTile = sV[cur];

        // S = Q * K^T for 32 keys (two 16x16 f32 tiles), K-dim = 128
        v8f s0, s1;
#pragma unroll
        for (int r = 0; r < 8; ++r) { s0[r] = 0.f; s1[r] = 0.f; }
#pragma unroll
        for (int kk = 0; kk < 4; ++kk) {
            v16bf b0 = load_b(kTile, 128, 0,  kk * 32);
            v16bf b1 = load_b(kTile, 128, 16, kk * 32);
            s0 = wmma_bf16(qa[kk], b0, s0);
            s1 = wmma_bf16(qa[kk], b1, s1);
        }

        // online softmax (per-row reduction over 16-lane halves)
        float scale[8];
#pragma unroll
        for (int r = 0; r < 8; ++r) {
            float v = fmaxf(s0[r], s1[r]);
            v = fmaxf(v, __shfl_xor(v, 1, 32));
            v = fmaxf(v, __shfl_xor(v, 2, 32));
            v = fmaxf(v, __shfl_xor(v, 4, 32));
            v = fmaxf(v, __shfl_xor(v, 8, 32));
            float mnew = fmaxf(mrow[r], v);
            scale[r] = exp2f((mrow[r] - mnew) * L2E);
            float p0 = exp2f((s0[r] - mnew) * L2E);
            float p1 = exp2f((s1[r] - mnew) * L2E);
            float rs = p0 + p1;
            rs += __shfl_xor(rs, 1, 32);
            rs += __shfl_xor(rs, 2, 32);
            rs += __shfl_xor(rs, 4, 32);
            rs += __shfl_xor(rs, 8, 32);
            lrow[r] = lrow[r] * scale[r] + rs;
            mrow[r] = mnew;
            int m = r + 8 * h;
            myP[m * 32 + (lane & 15)]      = f2bf(p0);   // C-layout -> LDS
            myP[m * 32 + 16 + (lane & 15)] = f2bf(p1);
        }
#pragma unroll
        for (int t = 0; t < 8; ++t)
            for (int r = 0; r < 8; ++r) acc[t][r] *= scale[r];

        // Reload P as an A fragment (16x32) from LDS
        FragAB pf;
        {
            const uint4* p = (const uint4*)(myP + (lane & 15) * 32 + h * 8);
            pf.q[0] = p[0];
            pf.q[1] = p[2];
        }
        // acc += P * V  (8 vc-tiles of 16, K = 32 keys) from LDS V tile
#pragma unroll
        for (int t = 0; t < 8; ++t) {
            v16bf vb = load_b(vTile, 32, t * 16, 0);
            acc[t] = wmma_bf16(pf.v, vb, acc[t]);
        }

        __syncthreads();                 // all waves done with cur before reuse
        cur = nxt;
    }

    unsigned short* Orow = Ob + (long long)b * N * VC;
    float inv[8];
#pragma unroll
    for (int r = 0; r < 8; ++r) inv[r] = 1.0f / lrow[r];
#pragma unroll
    for (int t = 0; t < 8; ++t)
        for (int r = 0; r < 8; ++r) {
            int m = r + 8 * h;
            Orow[(long long)(i0 + m) * VC + t * 16 + (lane & 15)] =
                f2bf(acc[t][r] * inv[r]);
        }
}

// ---------------------------------------------------------------------------
// Output projection + residual: out[b,c,n] = gamma*(Wo@O + bo) + x[b,c,n]
// ---------------------------------------------------------------------------
__global__ __launch_bounds__(128) void outproj_kernel(
    const unsigned short* __restrict__ Ob,   // [B][N][VC]
    const unsigned short* __restrict__ Wob,  // [C][VC]
    const float* __restrict__ bo,
    const float* __restrict__ x,
    const float* __restrict__ gamma,
    float* __restrict__ out,
    int B, int C, int N, int VC)
{
    int wave = (blockIdx.x * blockDim.x + threadIdx.x) >> 5;
    int lane = threadIdx.x & 31;
    int nTilesN = N / 64;
    int nTilesC = C / 16;
    int nt  = wave % nTilesN;
    int tmp = wave / nTilesN;
    int ct  = tmp % nTilesC;
    int b   = tmp / nTilesC;
    if (b >= B) return;
    int c0 = ct * 16, n0 = nt * 64;
    int h  = lane >> 4;
    float g = gamma[0];

    v8f acc[4];
    for (int r = 0; r < 8; ++r) {
        float bias = bo[c0 + r + 8 * h];
        for (int t = 0; t < 4; ++t) acc[t][r] = bias;
    }

    const unsigned short* Orow = Ob + (long long)b * N * VC;
    for (int k0 = 0; k0 < VC; k0 += 32) {
        v16bf a = load_a(Wob, VC, c0, k0);
        for (int t = 0; t < 4; ++t) {
            v16bf bf = load_b(Orow, VC, n0 + t * 16, k0);
            acc[t] = wmma_bf16(a, bf, acc[t]);
        }
    }

    for (int t = 0; t < 4; ++t)
        for (int r = 0; r < 8; ++r) {
            int cc = c0 + r + 8 * h;
            int n  = n0 + t * 16 + (lane & 15);
            long long idx = ((long long)b * C + cc) * N + n;
            out[idx] = g * acc[t][r] + x[idx];
        }
}

// ---------------------------------------------------------------------------
extern "C" void kernel_launch(void* const* d_in, const int* in_sizes, int n_in,
                              void* d_out, int out_size, void* d_ws, size_t ws_size,
                              hipStream_t stream) {
    const int B = 4, C = 256, N = 64 * 64, KC = 128, VC = 128;
    const float* x     = (const float*)d_in[0];
    const float* Wq    = (const float*)d_in[1];
    const float* bq    = (const float*)d_in[2];
    const float* Wk    = (const float*)d_in[3];
    const float* bk    = (const float*)d_in[4];
    const float* Wv    = (const float*)d_in[5];
    const float* bv    = (const float*)d_in[6];
    const float* Wo    = (const float*)d_in[7];
    const float* bo    = (const float*)d_in[8];
    const float* gamma = (const float*)d_in[9];
    float* out = (float*)d_out;

    char* ws = (char*)d_ws;
    size_t off = 0;
    unsigned short* xb    = (unsigned short*)(ws + off); off += (size_t)B * N * C * 2;
    unsigned short* Wcatb = (unsigned short*)(ws + off); off += (size_t)3 * KC * C * 2;
    unsigned short* Wob   = (unsigned short*)(ws + off); off += (size_t)C * VC * 2;
    unsigned short* Qb    = (unsigned short*)(ws + off); off += (size_t)B * N * KC * 2;
    unsigned short* Kb    = (unsigned short*)(ws + off); off += (size_t)B * N * KC * 2;
    unsigned short* Vt    = (unsigned short*)(ws + off); off += (size_t)B * VC * N * 2;
    unsigned short* Ob    = (unsigned short*)(ws + off);

    {
        long long total = (long long)B * N * C;
        pack_x_kernel<<<(int)((total + 255) / 256), 256, 0, stream>>>(x, xb, B, C, N);
    }
    {
        int total = 3 * KC * C + C * VC;
        pack_w_kernel<<<(total + 255) / 256, 256, 0, stream>>>(Wq, Wk, Wv, Wo,
                                                               Wcatb, Wob, KC, VC, C);
    }
    {
        int waves = B * (3 * KC / 16) * (N / 64);   // 6144
        qkv_kernel<<<waves / 4, 128, 0, stream>>>(xb, Wcatb, bq, bk, bv,
                                                  Qb, Kb, Vt, B, C, N, KC, VC);
    }
    {
        int blocks = B * (N / 64);                  // 256 blocks, 4 waves each
        attn_kernel<<<blocks, 128, 0, stream>>>(Qb, Kb, Vt, Ob, B, N, KC, VC);
    }
    {
        int waves = B * (C / 16) * (N / 64);        // 4096
        outproj_kernel<<<waves / 4, 128, 0, stream>>>(Ob, Wob, bo, x, gamma, out,
                                                      B, C, N, VC);
    }
}
